// MultiHeadAttentionBlock_50861002719813
// MI455X (gfx1250) — compile-verified, matrix operand order checked by reading, not run on hardware
//
#include <hip/hip_runtime.h>

// ---------------------------------------------------------------------------
// MultiHeadAttention for MI455X (gfx1250): fp16 WMMA pipeline + async-LDS
//   B=2, S=4096, D=1024, H=16, DK=64
//   1) Q = (q @ Wq^T) * 0.125  -> fp16, layout [b][h][s][dk]
//      K = (k @ Wk^T)          -> fp16, layout [b][h][s][dk]
//      Vt = (v @ Wv^T)^T       -> fp16, layout [b][h][dk][s]
//   2) fused flash attention (online softmax), fp16 WMMA, f32 accum,
//      K/Vt tiles double-buffered via global_load_async_to_lds_b128
//      -> AO fp16, layout [b][s][h*DK+dk]
//   3) out = AO @ Wo^T -> f32
// ---------------------------------------------------------------------------

typedef _Float16 v16h __attribute__((ext_vector_type(16)));
typedef _Float16 v8h  __attribute__((ext_vector_type(8)));
typedef _Float16 v4h  __attribute__((ext_vector_type(4)));
typedef float    v8f  __attribute__((ext_vector_type(8)));

static constexpr int S_  = 4096;
static constexpr int D_  = 1024;
static constexpr int H_  = 16;
static constexpr int DK_ = 64;
static constexpr int LDT = 72;   // padded LDS row stride (halfs): 144B, breaks bank conflicts

static __device__ __forceinline__ v8f wmma_f16(v16h a, v16h b, v8f c) {
  return __builtin_amdgcn_wmma_f32_16x16x32_f16(false, a, false, b, (short)0, c, false, false);
}

// Load one 16x32 fp16 fragment (A-style / N-major B-style) from LDS.
// lane L: row m = L%16, half hh = L/16; K runs {8hh+0..7} and {16+8hh+0..7}.
static __device__ __forceinline__ v16h ld_frag(const _Float16* p, int ld) {
  const int lane = threadIdx.x & 31;
  const int m  = lane & 15;
  const int hh = lane >> 4;
  const _Float16* q = p + m * ld + hh * 8;
  union { v16h v; v8h h[2]; } u;
  u.h[0] = *reinterpret_cast<const v8h*>(q);
  u.h[1] = *reinterpret_cast<const v8h*>(q + 16);
  return u.v;
}

// 64x64 f32 tile (global, stride sld) -> fp16 LDS tile (stride LDT). 128 threads.
static __device__ __forceinline__ void cvt_tile64(_Float16* dst, const float* src, int sld) {
#pragma unroll
  for (int i = 0; i < 8; ++i) {
    int idx = (int)threadIdx.x + i * 128;
    int r = idx >> 4;
    int c = (idx & 15) << 2;
    float4 f = *reinterpret_cast<const float4*>(src + r * sld + c);
    v4h hv = { (_Float16)f.x, (_Float16)f.y, (_Float16)f.z, (_Float16)f.w };
    *reinterpret_cast<v4h*>(dst + r * LDT + c) = hv;
  }
}

// 64x64 fp16 tile (global, stride sld) -> fp16 LDS tile (stride LDT). 128 threads.
static __device__ __forceinline__ void cp_tile64(_Float16* dst, const _Float16* src, int sld) {
#pragma unroll
  for (int i = 0; i < 4; ++i) {
    int idx = (int)threadIdx.x + i * 128;
    int r = idx >> 3;
    int c = (idx & 7) << 3;
    *reinterpret_cast<uint4*>(dst + r * LDT + c) =
        *reinterpret_cast<const uint4*>(src + r * sld + c);
  }
}

// 64x64 fp16 tile via CDNA5 async copy: global -> LDS, no VGPR round trip.
// GVS addressing: mem = SGPR64(base) + VGPR32(byte offset); LDS dest byte
// address = low 32 bits of the generic shared pointer (ADDR[31:0] = LDS off).
// Tracked by ASYNCcnt; caller must s_wait_asynccnt before consuming.
static __device__ __forceinline__ void async_tile64(_Float16* dst, const _Float16* src_base, int sld) {
#pragma unroll
  for (int i = 0; i < 4; ++i) {
    int idx = (int)threadIdx.x + i * 128;
    int r = idx >> 3;
    int c = (idx & 7) << 3;
    unsigned lds_addr = (unsigned)(uintptr_t)(dst + r * LDT + c);
    unsigned goff = (unsigned)((r * sld + c) * 2);  // bytes
    asm volatile("global_load_async_to_lds_b128 %0, %1, %2"
                 :: "v"(lds_addr), "v"(goff), "s"(src_base)
                 : "memory");
  }
}

static __device__ __forceinline__ void wait_async0() {
  asm volatile("s_wait_asynccnt 0x0" ::: "memory");
}

// ---------------------------------------------------------------------------
// Projection GEMM: C[64x64 tile] = X @ W^T (+ layout-specific epilogue)
//   MODE 0: Q  (scale 0.125, head layout [b][h][s][dk], fp16 out)
//   MODE 1: K  (head layout, fp16 out)
//   MODE 2: V  (transposed head layout [b][h][dk][s], fp16 out)
//   MODE 3: final out = AO(fp16) @ Wo^T -> f32 row-major [B*S, D]
// grid: (M/64, D/64), block: 128 (4 waves); wave w owns rows 16w..16w+15.
// ---------------------------------------------------------------------------
template <int MODE>
__global__ __launch_bounds__(128) void mha_proj_kernel(
    const float* __restrict__ Xf, const _Float16* __restrict__ Xh,
    const float* __restrict__ W,
    _Float16* __restrict__ dstH, float* __restrict__ dstF) {
  __shared__ _Float16 At[64 * LDT];
  __shared__ _Float16 Bt[64 * LDT];

  const int m0 = blockIdx.x * 64;
  const int n0 = blockIdx.y * 64;
  const int wave = threadIdx.x >> 5;
  const int lane = threadIdx.x & 31;
  const int hh = lane >> 4;
  const int col = lane & 15;

  const v8f zf = {0.f, 0.f, 0.f, 0.f, 0.f, 0.f, 0.f, 0.f};
  v8f acc[4] = {zf, zf, zf, zf};

  for (int k0 = 0; k0 < D_; k0 += 64) {
    __syncthreads();
    if (MODE < 3) {
      cvt_tile64(At, Xf + (size_t)m0 * D_ + k0, D_);
    } else {
      cp_tile64(At, Xh + (size_t)m0 * D_ + k0, D_);
    }
    cvt_tile64(Bt, W + (size_t)n0 * D_ + k0, D_);
    // prefetch next K-step tiles into cache (global_prefetch_b8)
    if (k0 + 64 < D_) {
      const int r = (int)threadIdx.x >> 1;        // 0..63
      const int cq = ((int)threadIdx.x & 1) << 5; // 0 or 32
      if (MODE < 3) __builtin_prefetch(Xf + (size_t)(m0 + r) * D_ + k0 + 64 + cq, 0, 3);
      else          __builtin_prefetch(Xh + (size_t)(m0 + r) * D_ + k0 + 64 + cq, 0, 3);
      __builtin_prefetch(W + (size_t)(n0 + r) * D_ + k0 + 64 + cq, 0, 3);
    }
    __syncthreads();

    v16h a0 = ld_frag(&At[(wave * 16) * LDT], LDT);
    v16h a1 = ld_frag(&At[(wave * 16) * LDT + 32], LDT);
#pragma unroll
    for (int st = 0; st < 4; ++st) {
      v16h b0 = ld_frag(&Bt[(st * 16) * LDT], LDT);
      v16h b1 = ld_frag(&Bt[(st * 16) * LDT + 32], LDT);
      acc[st] = wmma_f16(a0, b0, acc[st]);
      acc[st] = wmma_f16(a1, b1, acc[st]);
    }
  }

  // Epilogue: C row = j + 8*hh, C col = col (per 16-wide subtile st)
#pragma unroll
  for (int st = 0; st < 4; ++st) {
#pragma unroll
    for (int j = 0; j < 8; ++j) {
      float v = acc[st][j];
      if (MODE == 0) v *= 0.125f;  // 1/sqrt(DK)
      const int gm = m0 + wave * 16 + hh * 8 + j;  // global row in [0, B*S)
      const int gn = n0 + st * 16 + col;           // global col in [0, D)
      if (MODE == 3) {
        dstF[(size_t)gm * D_ + gn] = v;
      } else {
        const int b = gm >> 12;           // / S
        const int s = gm & (S_ - 1);
        const int hd = gn >> 6;           // head
        const int dk = gn & (DK_ - 1);
        if (MODE == 2) {
          // Vt[b][h][dk][s]
          dstH[(((size_t)(b * H_ + hd)) * DK_ + dk) * S_ + s] = (_Float16)v;
        } else {
          // Q/K [b][h][s][dk]
          dstH[(((size_t)(b * H_ + hd)) * S_ + s) * DK_ + dk] = (_Float16)v;
        }
      }
    }
  }
}

// ---------------------------------------------------------------------------
// Fused flash attention. grid = B*H*(S/64), block = 128 (4 waves).
// Each wave: 16 query rows x DK=64; loop over 64-key tiles with online
// softmax. K/Vt tiles double-buffered via async LDS loads.
// ---------------------------------------------------------------------------
__global__ __launch_bounds__(128) void mha_flash_kernel(
    const _Float16* __restrict__ Qh, const _Float16* __restrict__ Kh,
    const _Float16* __restrict__ Vtg, _Float16* __restrict__ AO) {
  __shared__ _Float16 Qt[64 * LDT];
  __shared__ _Float16 Kt[2][64 * LDT];
  __shared__ _Float16 Vtl[2][64 * LDT];
  __shared__ _Float16 Pt[64 * LDT];

  const int bh = blockIdx.x >> 6;          // b*H + h
  const int q0 = (blockIdx.x & 63) * 64;   // query tile base
  const int wave = threadIdx.x >> 5;
  const int lane = threadIdx.x & 31;
  const int hh = lane >> 4;
  const int col = lane & 15;

  const _Float16* Kbase  = Kh  + (size_t)bh * S_ * DK_;
  const _Float16* Vtbase = Vtg + (size_t)bh * DK_ * S_;

  // Kick off async loads of the first K/Vt tiles, then stage Q while in flight.
  async_tile64(Kt[0],  Kbase, DK_);
  async_tile64(Vtl[0], Vtbase, S_);

  cp_tile64(Qt, Qh + ((size_t)bh * S_ + q0) * DK_, DK_);
  __syncthreads();
  const v16h qf0 = ld_frag(&Qt[(wave * 16) * LDT], LDT);
  const v16h qf1 = ld_frag(&Qt[(wave * 16) * LDT + 32], LDT);

  const v8f zf = {0.f, 0.f, 0.f, 0.f, 0.f, 0.f, 0.f, 0.f};
  v8f accO[4] = {zf, zf, zf, zf};
  float rmax[8], rsum[8];
#pragma unroll
  for (int j = 0; j < 8; ++j) { rmax[j] = -__builtin_inff(); rsum[j] = 0.f; }

  _Float16* Pw = &Pt[(wave * 16) * LDT];

  for (int kt = 0; kt < S_ / 64; ++kt) {
    const int cur = kt & 1;
    wait_async0();        // this wave's async writes (tile kt) are in LDS
    __syncthreads();      // everyone's tile-kt writes visible; tile kt-1 reads done
    if (kt + 1 < S_ / 64) {
      async_tile64(Kt[1 - cur],  Kbase  + (size_t)(kt + 1) * 64 * DK_, DK_);
      async_tile64(Vtl[1 - cur], Vtbase + (size_t)(kt + 1) * 64,       S_);
    }

    // scores: 16 queries x 64 keys (4 subtiles), f32
    v8f sc[4];
#pragma unroll
    for (int st = 0; st < 4; ++st) {
      v16h kf0 = ld_frag(&Kt[cur][(st * 16) * LDT], LDT);
      v16h kf1 = ld_frag(&Kt[cur][(st * 16) * LDT + 32], LDT);
      sc[st] = wmma_f16(qf0, kf0, zf);
      sc[st] = wmma_f16(qf1, kf1, sc[st]);
    }

    // online softmax per row (row = hh*8 + j), reduce across the 16-lane half
#pragma unroll
    for (int j = 0; j < 8; ++j) {
      float tmax = fmaxf(fmaxf(sc[0][j], sc[1][j]), fmaxf(sc[2][j], sc[3][j]));
#pragma unroll
      for (int m = 1; m < 16; m <<= 1) tmax = fmaxf(tmax, __shfl_xor(tmax, m, 32));
      const float mnew = fmaxf(rmax[j], tmax);
      const float alpha = __expf(rmax[j] - mnew);
      rmax[j] = mnew;
      float psum = 0.f;
#pragma unroll
      for (int st = 0; st < 4; ++st) {
        float p = __expf(sc[st][j] - mnew);
        psum += p;
        Pw[(hh * 8 + j) * LDT + st * 16 + col] = (_Float16)p;
      }
#pragma unroll
      for (int m = 1; m < 16; m <<= 1) psum += __shfl_xor(psum, m, 32);
      rsum[j] = rsum[j] * alpha + psum;
#pragma unroll
      for (int st = 0; st < 4; ++st) accO[st][j] *= alpha;
    }

    // P (16x64) @ V (64 keys x 64 dk) via Vt fragments (N-major rows of Vt)
    const v16h pf0 = ld_frag(Pw, LDT);
    const v16h pf1 = ld_frag(Pw + 32, LDT);
#pragma unroll
    for (int st = 0; st < 4; ++st) {
      v16h vf0 = ld_frag(&Vtl[cur][(st * 16) * LDT], LDT);
      v16h vf1 = ld_frag(&Vtl[cur][(st * 16) * LDT + 32], LDT);
      accO[st] = wmma_f16(pf0, vf0, accO[st]);
      accO[st] = wmma_f16(pf1, vf1, accO[st]);
    }
  }

  // epilogue: AO[b][s][h*64 + dk], fp16
  const int b = bh >> 4;
  const int hd = bh & 15;
#pragma unroll
  for (int st = 0; st < 4; ++st) {
#pragma unroll
    for (int j = 0; j < 8; ++j) {
      const float v = accO[st][j] / rsum[j];
      const int s = q0 + wave * 16 + hh * 8 + j;
      AO[((size_t)b * S_ + s) * D_ + hd * DK_ + st * 16 + col] = (_Float16)v;
    }
  }
}

// ---------------------------------------------------------------------------
extern "C" void kernel_launch(void* const* d_in, const int* in_sizes, int n_in,
                              void* d_out, int out_size, void* d_ws, size_t ws_size,
                              hipStream_t stream) {
  const float* q  = (const float*)d_in[0];
  const float* k  = (const float*)d_in[1];
  const float* v  = (const float*)d_in[2];
  // d_in[3] = mask : no-op in the reference
  const float* Wq = (const float*)d_in[4];
  const float* Wk = (const float*)d_in[5];
  const float* Wv = (const float*)d_in[6];
  const float* Wo = (const float*)d_in[7];

  const size_t E = (size_t)2 * H_ * S_ * DK_;  // 8,388,608 halfs = 16 MB
  _Float16* Qh = (_Float16*)d_ws;
  _Float16* Kh = Qh + E;
  _Float16* Vt = Kh + E;
  _Float16* AO = Vt + E;

  const dim3 gp(128, 16);   // (B*S/64, D/64)
  const dim3 blk(128);

  mha_proj_kernel<0><<<gp, blk, 0, stream>>>(q, nullptr, Wq, Qh, nullptr);
  mha_proj_kernel<1><<<gp, blk, 0, stream>>>(k, nullptr, Wk, Kh, nullptr);
  mha_proj_kernel<2><<<gp, blk, 0, stream>>>(v, nullptr, Wv, Vt, nullptr);

  mha_flash_kernel<<<dim3(2 * H_ * (S_ / 64)), blk, 0, stream>>>(Qh, Kh, Vt, AO);

  mha_proj_kernel<3><<<gp, blk, 0, stream>>>(nullptr, AO, Wo, nullptr, (float*)d_out);
}
